// Mamba2ForSentimentAnalysis_56212531970147
// MI455X (gfx1250) — matrix operand order repaired and checked by reading, not compiled, not run on hardware
//
#include <hip/hip_runtime.h>
#include <hip/hip_bf16.h>

typedef __attribute__((ext_vector_type(16))) _Float16 v16h;
typedef __attribute__((ext_vector_type(8)))  _Float16 v8h;
typedef __attribute__((ext_vector_type(8)))  float    v8f;

#define D_MODEL 128
#define D_STATE 16
#define D_INNER 256
#define NB      16
#define LSEQ    2048
#define MTOT    (NB * LSEQ)   // 32768 tokens
#define NPROJ   48            // x_proj rows padded 40 -> 48 for 16x16 tiles

// ---------------------------------------------------------------------------
// Weight precision prep: fp32 -> fp16 copies of in_proj_w (512x128) and the
// zero-padded x_proj_w (48x256). Done per launch (deterministic).
// ---------------------------------------------------------------------------
__global__ void prep_weights_kernel(const float* __restrict__ inpw,
                                    const float* __restrict__ xpw,
                                    _Float16* __restrict__ inpw16,
                                    _Float16* __restrict__ xpw16) {
    int idx = blockIdx.x * blockDim.x + threadIdx.x;
    const int n1 = 2 * D_INNER * D_MODEL;              // 65536
    if (idx < n1) {
        inpw16[idx] = (_Float16)inpw[idx];
    }
    int j = idx - n1;
    if (j >= 0 && j < NPROJ * D_INNER) {               // 12288
        int n = j >> 8, k = j & 255;
        xpw16[j] = (n < 40) ? (_Float16)xpw[n * D_INNER + k] : (_Float16)0.f;
    }
}

// ---------------------------------------------------------------------------
// Embedding gather, emitted directly as fp16 GEMM operand A (MTOT x 128).
// ---------------------------------------------------------------------------
__global__ void embed_kernel(const int* __restrict__ ids,
                             const float* __restrict__ emb,
                             _Float16* __restrict__ x16) {
    int idx = blockIdx.x * blockDim.x + threadIdx.x;   // MTOT*128 threads
    int m = idx >> 7, d = idx & 127;
    x16[idx] = (_Float16)emb[(size_t)ids[m] * D_MODEL + d];
}

// ---------------------------------------------------------------------------
// WMMA GEMM: C[M x N] = A[M x K] * Bw^T  (Bw stored N x K row-major).
// One wave per 16x16 C tile. Fragment layouts per CDNA5 ISA 7.12.2:
//   A 16x32 f16 : lane<16 row M=lane, elems 0..7 = K+0..7, 8..15 = K+16..23
//                 lane>=16 same row,  elems 0..7 = K+8..15, 8..15 = K+24..31
//   B 32x16 f16 : col N = lane&15, same K order as A (elems 0..7 = K+0..7 / +8..15, 8..15 = K+16..23 / +24..31)
//   D 16x16 f32 : VGPR v -> row M = v (+8 for lanes>=16), col N = lane&15
// ---------------------------------------------------------------------------
__global__ __launch_bounds__(256) void gemm_wmma_f16_kernel(
        const _Float16* __restrict__ A, const _Float16* __restrict__ Bw,
        float* __restrict__ C, int ntiles, int K, int N) {
    int wid  = (blockIdx.x * blockDim.x + threadIdx.x) >> 5;
    int lane = threadIdx.x & 31;
    int mtile = wid / ntiles;
    int ntile = wid - mtile * ntiles;
    int hi16  = (lane >> 4) & 1;
    int mrow  = (mtile << 4) + (lane & 15);
    int ncol  = (ntile << 4) + (lane & 15);

    const _Float16* aBase = A  + (size_t)mrow * K + (hi16 ? 8  : 0);
    const _Float16* bBase = Bw + (size_t)ncol * K + (hi16 ? 8  : 0);

    v8f acc = {};
    for (int kk = 0; kk < K; kk += 32) {
        v8h alo = *(const v8h*)(aBase + kk);        // 16B load
        v8h ahi = *(const v8h*)(aBase + kk + 16);   // 16B load
        v16h a;
#pragma unroll
        for (int i = 0; i < 8; ++i) { a[i] = alo[i]; a[i + 8] = ahi[i]; }
        v8h blo = *(const v8h*)(bBase + kk);        // 16B load
        v8h bhi = *(const v8h*)(bBase + kk + 16);   // 16B load
        v16h b;
#pragma unroll
        for (int i = 0; i < 8; ++i) { b[i] = blo[i]; b[i + 8] = bhi[i]; }
        acc = __builtin_amdgcn_wmma_f32_16x16x32_f16(
                  false, a, false, b, (short)0, acc, false, false);
    }

#pragma unroll
    for (int v = 0; v < 8; ++v) {
        int row = (mtile << 4) + v + (hi16 ? 8 : 0);
        C[(size_t)row * N + ncol] = acc[v];
    }
}

// ---------------------------------------------------------------------------
// Depthwise causal conv (kernel 4) + bias + SiLU. Reads xc = xz[:, :256].
// Writes fp32 (scan input) and fp16 (x_proj GEMM operand).
// ---------------------------------------------------------------------------
__global__ void conv_silu_kernel(const float* __restrict__ xz,
                                 const float* __restrict__ cw,
                                 const float* __restrict__ cb,
                                 float* __restrict__ xc32,
                                 _Float16* __restrict__ xc16) {
    int idx = blockIdx.x * blockDim.x + threadIdx.x;   // MTOT*256 threads
    int m = idx >> 8, d = idx & 255;
    int b = m >> 11, l = m & (LSEQ - 1);
    float acc = cb[d];
#pragma unroll
    for (int k = 0; k < 4; ++k) {
        int lsrc = l + k - 3;
        if (lsrc >= 0)
            acc += cw[d * 4 + k] * xz[(size_t)((b << 11) + lsrc) * 512 + d];
    }
    float s = acc / (1.f + __expf(-acc));              // SiLU
    xc32[idx] = s;
    xc16[idx] = (_Float16)s;
}

// ---------------------------------------------------------------------------
// delta = softplus(dt @ dt_proj_w^T + b). dt = dbc[:, 0:8] (K=8, scalar VALU).
// ---------------------------------------------------------------------------
__global__ void dt_kernel(const float* __restrict__ dbc,
                          const float* __restrict__ dtw,
                          const float* __restrict__ dtb,
                          float* __restrict__ delta) {
    int idx = blockIdx.x * blockDim.x + threadIdx.x;   // MTOT*256 threads
    int m = idx >> 8, d = idx & 255;
    const float* row = dbc + (size_t)m * NPROJ;
    float acc = dtb[d];
#pragma unroll
    for (int r = 0; r < 8; ++r) acc += row[r] * dtw[d * 8 + r];
    delta[idx] = (acc > 20.f) ? acc : log1pf(__expf(acc));
}

// ---------------------------------------------------------------------------
// Selective scan: one thread per (b,d) recurrence lane; h[16] lives in VGPRs,
// per-step B/C (shared across d within a batch) broadcast through LDS.
// This is the serial critical path — 2048 steps of 16 exp+fma per lane.
// ---------------------------------------------------------------------------
__global__ __launch_bounds__(256) void scan_kernel(
        const float* __restrict__ delta, const float* __restrict__ xc32,
        const float* __restrict__ dbc,   const float* __restrict__ A_log,
        float* __restrict__ ys) {
    int b = blockIdx.x, d = threadIdx.x;
    __shared__ float Bs[D_STATE], Cs[D_STATE];
    float Av[D_STATE], h[D_STATE];
#pragma unroll
    for (int n = 0; n < D_STATE; ++n) {
        Av[n] = -__expf(A_log[d * D_STATE + n]);
        h[n]  = 0.f;
    }
    for (int l = 0; l < LSEQ; ++l) {
        size_t m = (size_t)b * LSEQ + l;
        if (d < 16)      Bs[d]      = dbc[m * NPROJ + 8  + d];
        else if (d < 32) Cs[d - 16] = dbc[m * NPROJ + 24 + (d - 16)];
        __syncthreads();
        float dt = delta[m * 256 + d];
        float xv = xc32[m * 256 + d];
        float y = 0.f;
#pragma unroll
        for (int n = 0; n < D_STATE; ++n) {
            float dA = __expf(dt * Av[n]);
            h[n] = dA * h[n] + dt * Bs[n] * xv;
            y += h[n] * Cs[n];
        }
        ys[m * 256 + d] = y;
        __syncthreads();
    }
}

// ---------------------------------------------------------------------------
// Gating + mean over L, fused: ybar[b,d] = mean_l((ys + xc*D) * silu(z)).
// (out_proj commutes with the mean, so it moves to the tiny head kernel.)
// ---------------------------------------------------------------------------
__global__ __launch_bounds__(256) void gate_reduce_kernel(
        const float* __restrict__ ys, const float* __restrict__ xc32,
        const float* __restrict__ xz, const float* __restrict__ Dv,
        float* __restrict__ ybar) {
    int b = blockIdx.x, d = threadIdx.x;
    float Dd = Dv[d];
    float sum = 0.f;
    for (int l = 0; l < LSEQ; ++l) {
        size_t m = (size_t)b * LSEQ + l;
        float z  = xz[m * 512 + 256 + d];
        float sz = z / (1.f + __expf(-z));
        sum += (ys[m * 256 + d] + xc32[m * 256 + d] * Dd) * sz;
    }
    ybar[b * 256 + d] = sum * (1.f / LSEQ);
}

// ---------------------------------------------------------------------------
// Head: logits = (ybar @ out_proj^T) @ cls_w^T + cls_b.  16x2 outputs.
// ---------------------------------------------------------------------------
__global__ __launch_bounds__(128) void head_kernel(
        const float* __restrict__ ybar, const float* __restrict__ opw,
        const float* __restrict__ clsw, const float* __restrict__ clsb,
        float* __restrict__ out) {
    __shared__ float xb[D_MODEL];
    int b = blockIdx.x, dm = threadIdx.x;
    float acc = 0.f;
    for (int e = 0; e < D_INNER; ++e)
        acc += ybar[b * 256 + e] * opw[dm * 256 + e];
    xb[dm] = acc;
    __syncthreads();
    if (dm < 2) {
        float lg = clsb[dm];
        for (int k = 0; k < D_MODEL; ++k) lg += xb[k] * clsw[dm * 128 + k];
        out[b * 2 + dm] = lg;
    }
}

// ---------------------------------------------------------------------------
extern "C" void kernel_launch(void* const* d_in, const int* in_sizes, int n_in,
                              void* d_out, int out_size, void* d_ws, size_t ws_size,
                              hipStream_t stream) {
    const int*   ids  = (const int*)  d_in[0];
    const float* emb  = (const float*)d_in[1];
    const float* inpw = (const float*)d_in[2];
    const float* cw   = (const float*)d_in[3];
    const float* cb   = (const float*)d_in[4];
    const float* xpw  = (const float*)d_in[5];
    const float* dtw  = (const float*)d_in[6];
    const float* dtb  = (const float*)d_in[7];
    const float* Alog = (const float*)d_in[8];
    const float* Dv   = (const float*)d_in[9];
    const float* opw  = (const float*)d_in[10];
    const float* clsw = (const float*)d_in[11];
    const float* clsb = (const float*)d_in[12];
    float* out = (float*)d_out;

    char* ws = (char*)d_ws;
    size_t off = 0;
    auto take = [&](size_t bytes) -> char* {
        char* p = ws + off;
        off = (off + bytes + 255) & ~(size_t)255;
        return p;
    };
    _Float16* xf16   = (_Float16*)take((size_t)MTOT * D_MODEL * 2);   //  8 MB
    _Float16* inpw16 = (_Float16*)take((size_t)512 * 128 * 2);        //  128 KB
    _Float16* xpw16  = (_Float16*)take((size_t)NPROJ * D_INNER * 2);  //  24 KB
    float*    xz     = (float*)   take((size_t)MTOT * 512 * 4);       // 64 MB
    float*    xc32   = (float*)   take((size_t)MTOT * 256 * 4);       // 32 MB
    _Float16* xc16   = (_Float16*)take((size_t)MTOT * 256 * 2);       // 16 MB
    float*    dbc    = (float*)   take((size_t)MTOT * NPROJ * 4);     //  6 MB
    float*    delta  = (float*)   take((size_t)MTOT * 256 * 4);       // 32 MB
    float*    ys     = (float*)   take((size_t)MTOT * 256 * 4);       // 32 MB
    float*    ybar   = (float*)   take((size_t)NB * 256 * 4);         // 16 KB

    // 1) fp16 weight prep (65536 + 12288 elements)
    prep_weights_kernel<<<(77824 + 255) / 256, 256, 0, stream>>>(inpw, xpw, inpw16, xpw16);
    // 2) embedding gather -> fp16 A operand
    embed_kernel<<<(MTOT * D_MODEL) / 256, 256, 0, stream>>>(ids, emb, xf16);
    // 3) in_proj: [32768x128] x [128x512] -> xz ; 2048*32 tiles / 8 waves
    gemm_wmma_f16_kernel<<<(2048 * 32) / 8, 256, 0, stream>>>(xf16, inpw16, xz, 32, 128, 512);
    // 4) depthwise causal conv + SiLU
    conv_silu_kernel<<<(MTOT * 256) / 256, 256, 0, stream>>>(xz, cw, cb, xc32, xc16);
    // 5) x_proj: [32768x256] x [256x48] -> dbc ; 2048*3 tiles / 8 waves
    gemm_wmma_f16_kernel<<<(2048 * 3) / 8, 256, 0, stream>>>(xc16, xpw16, dbc, 3, 256, NPROJ);
    // 6) delta = softplus(dt_proj)
    dt_kernel<<<(MTOT * 256) / 256, 256, 0, stream>>>(dbc, dtw, dtb, delta);
    // 7) selective scan (serial critical path)
    scan_kernel<<<NB, 256, 0, stream>>>(delta, xc32, dbc, Alog, ys);
    // 8) gating + mean over L
    gate_reduce_kernel<<<NB, 256, 0, stream>>>(ys, xc32, xz, Dv, ybar);
    // 9) out_proj (folded past the mean) + classifier
    head_kernel<<<NB, 128, 0, stream>>>(ybar, opw, clsw, clsb, out);
}
